// MultiHeadAttention_3229815406804
// MI455X (gfx1250) — matrix operand order repaired and checked by reading, not compiled, not run on hardware
//
#include <hip/hip_runtime.h>

typedef _Float16 half_t;
typedef __attribute__((ext_vector_type(16))) _Float16 v16h;
typedef __attribute__((ext_vector_type(8)))  _Float16 v8h;
typedef __attribute__((ext_vector_type(8)))  float    v8f;

#define B_   2
#define S_   2048
#define D_   768
#define H_   12
#define DH_  64

// ---------------------------------------------------------------------------
// WMMA wrapper: D = A(16x32 f16) * B(32x16 f16) + C(16x16 f32), wave32.
// ---------------------------------------------------------------------------
__device__ __forceinline__ v8f wmma_f16(v16h a, v16h b, v8f c) {
  return __builtin_amdgcn_wmma_f32_16x16x32_f16(
      /*neg_a=*/false, a, /*neg_b=*/false, b,
      /*c_mod=*/(short)0, c, /*reuse_a=*/false, /*reuse_b=*/false);
}

// ---------------------------------------------------------------------------
// Async copy: 16B per lane, memory -> LDS, tracked with ASYNCcnt (ISA 10.x /
// async_tensor doc). vdst is the per-lane LDS byte offset in a VGPR; the flat
// address of a __shared__ object carries the LDS offset in its low 32 bits
// (LDS aperture truncates addr[31:0]).
// ---------------------------------------------------------------------------
__device__ __forceinline__ void async_copy_b128(unsigned lds_off, const void* gaddr) {
  asm volatile("global_load_async_to_lds_b128 %0, %1, off"
               :: "v"(lds_off), "v"(gaddr)
               : "memory");
}
__device__ __forceinline__ void wait_asynccnt0() {
  asm volatile("s_wait_asynccnt 0x0" ::: "memory");
}
__device__ __forceinline__ void wait_dscnt0() {
  asm volatile("s_wait_dscnt 0x0" ::: "memory");
}

// ---------------------------------------------------------------------------
// A operand (16x32, MxK) from row-major f16 with leading dim ld.
// Lane = half*16 + m; elems 0..7 -> K = 8*half + e; elems 8..15 -> 16+8*half+e.
// ---------------------------------------------------------------------------
__device__ __forceinline__ v16h load_a(const half_t* A, int ld, int m0, int k0) {
  const int lane   = threadIdx.x & 31;
  const int halfid = lane >> 4;
  const int m      = lane & 15;
  const half_t* base = A + (size_t)(m0 + m) * ld + k0 + 8 * halfid;
  v8h lo = *(const v8h*)(base);
  v8h hi = *(const v8h*)(base + 16);
  v16h a;
#pragma unroll
  for (int i = 0; i < 8; ++i) { a[i] = lo[i]; a[8 + i] = hi[i]; }
  return a;
}

// ---------------------------------------------------------------------------
// B operand (32x16, KxN) where B[k][n] = P[(n0+n)*ld + k0 + k].
// Lane = half*16 + n; elems 0..7 -> K = 8*half + e; elems 8..15 -> 16+8*half+e.
// ---------------------------------------------------------------------------
__device__ __forceinline__ v16h load_b(const half_t* P, int ld, int n0, int k0) {
  const int lane = threadIdx.x & 31;
  const int n    = lane & 15;
  const int koff = (lane >> 4) * 8;
  const half_t* base = P + (size_t)(n0 + n) * ld + k0 + koff;
  v8h lo = *(const v8h*)(base);
  v8h hi = *(const v8h*)(base + 16);
  v16h b;
#pragma unroll
  for (int i = 0; i < 8; ++i) { b[i] = lo[i]; b[8 + i] = hi[i]; }
  return b;
}

// ---------------------------------------------------------------------------
// f32 -> f16 conversion (grid-stride)
// ---------------------------------------------------------------------------
__global__ void cvt_f32_f16(const float* __restrict__ in,
                            half_t* __restrict__ out, int n) {
  int i = blockIdx.x * blockDim.x + threadIdx.x;
  int stride = gridDim.x * blockDim.x;
  for (; i < n; i += stride) out[i] = (half_t)in[i];
}

// ---------------------------------------------------------------------------
// QKV projection: out = x @ W^T, split heads. 1 wave computes 32 rows x 64
// cols (8 accumulators); each weight B-tile is reused by two A-tiles, so the
// inner loop runs 12 b128 loads per 8 WMMAs.
// transpose_out=0: out[b,H,s,Dh] (Q,K);  =1: out[b,H,Dh,s] (V transposed).
// ---------------------------------------------------------------------------
__global__ __launch_bounds__(32) void qkv_proj_kernel(
    const half_t* __restrict__ xh, const half_t* __restrict__ Wh,
    half_t* __restrict__ out, int transpose_out) {
  const int m0 = blockIdx.x * 32;
  const int h  = blockIdx.y;

  v8f acc[2][4] = {};
  for (int kk = 0; kk < D_; kk += 32) {
    v16h a0 = load_a(xh, D_, m0,      kk);
    v16h a1 = load_a(xh, D_, m0 + 16, kk);
#pragma unroll
    for (int nt = 0; nt < 4; ++nt) {
      v16h b = load_b(Wh, D_, h * DH_ + nt * 16, kk);
      acc[0][nt] = wmma_f16(a0, b, acc[0][nt]);
      acc[1][nt] = wmma_f16(a1, b, acc[1][nt]);
    }
  }

  const int lane   = threadIdx.x & 31;
  const int halfid = lane >> 4;
  const int n      = lane & 15;
#pragma unroll
  for (int mt = 0; mt < 2; ++mt) {
#pragma unroll
    for (int nt = 0; nt < 4; ++nt) {
#pragma unroll
      for (int r = 0; r < 8; ++r) {
        const int flat = m0 + mt * 16 + r + 8 * halfid;  // token in [0, B*S)
        const int bb   = flat >> 11;
        const int srow = flat & (S_ - 1);
        const int dh   = nt * 16 + n;
        const half_t val = (half_t)acc[mt][nt][r];
        if (transpose_out)
          out[((size_t)(bb * H_ + h) * DH_ + dh) * S_ + srow] = val;
        else
          out[((size_t)(bb * H_ + h) * S_ + srow) * DH_ + dh] = val;
      }
    }
  }
}

// ---------------------------------------------------------------------------
// Flash attention (causal), 1 wave per (b, h, 32-row q-tile).
// Per 32-token K-block: K (contiguous 4KB) and V (transposed, 64x32) are
// staged to LDS with global_load_async_to_lds_b128 (ASYNCcnt), then the B
// operands are ds_read from LDS. 8 WMMAs for Q*K^T, online softmax in the
// C/D layout, P reshaped through LDS, 8 WMMAs for P*V.
// ---------------------------------------------------------------------------
__global__ __launch_bounds__(32) void flash_attn_kernel(
    const half_t* __restrict__ Q, const half_t* __restrict__ K,
    const half_t* __restrict__ Vt, half_t* __restrict__ ctx) {
  const int qt = blockIdx.x;   // 0..63  (32 query rows per tile)
  const int h  = blockIdx.y;   // 0..11
  const int bb = blockIdx.z;   // 0..1

  const half_t* Qbh = Q  + (size_t)(bb * H_ + h) * S_ * DH_;
  const half_t* Kbh = K  + (size_t)(bb * H_ + h) * S_ * DH_;
  const half_t* Vbh = Vt + (size_t)(bb * H_ + h) * DH_ * S_;

  __shared__ half_t ldsK[32 * 64];   // [token][dh]   4KB
  __shared__ half_t ldsV[64 * 32];   // [dh][token]   4KB
  __shared__ half_t ldsP[32 * 32];   // P reshape     2KB
  const unsigned kBase = (unsigned)(size_t)(void*)ldsK;
  const unsigned vBase = (unsigned)(size_t)(void*)ldsV;

  const int lane   = threadIdx.x & 31;
  const int halfid = lane >> 4;
  const int n      = lane & 15;

  v16h aq[2][2];
#pragma unroll
  for (int qs = 0; qs < 2; ++qs) {
    aq[qs][0] = load_a(Qbh, DH_, qt * 32 + qs * 16, 0);
    aq[qs][1] = load_a(Qbh, DH_, qt * 32 + qs * 16, 32);
  }

  v8f o[2][4] = {};
  float m_i[2][8], l_i[2][8];
#pragma unroll
  for (int qs = 0; qs < 2; ++qs)
#pragma unroll
    for (int r = 0; r < 8; ++r) { m_i[qs][r] = -3.0e38f; l_i[qs][r] = 0.0f; }

  const float scale = 0.125f;     // 1/sqrt(64)
  const int   nblk  = qt + 1;     // 32-token K-blocks (rows reach qt*32+31)

  for (int kb = 0; kb < nblk; ++kb) {
    const int tok0 = kb * 32;

    // ---- stage K and V blocks into LDS (async, 16B per lane per op) ----
    wait_dscnt0();   // previous block's ds_reads must land before overwrite
    {
      const char* gK = (const char*)(Kbh + (size_t)tok0 * DH_);  // 4KB contig
#pragma unroll
      for (int c = 0; c < 8; ++c)
        async_copy_b128(kBase + (unsigned)(lane * 16 + c * 512),
                        gK + lane * 16 + c * 512);
      const int vrow  = lane >> 2;        // 0..7
      const int vcol8 = (lane & 3) * 8;   // half-elements within 64B row seg
#pragma unroll
      for (int c = 0; c < 8; ++c) {
        const int dh = c * 8 + vrow;      // 0..63
        async_copy_b128(vBase + (unsigned)((dh * 32 + vcol8) * 2),
                        (const void*)(Vbh + (size_t)dh * S_ + tok0 + vcol8));
      }
    }
    wait_asynccnt0();
    __syncthreads();   // single-wave workgroup: compiler fence only

    // ---- S = Q * K^T  (8 WMMAs, B-tiles from LDS reused by both q-subtiles)
    v8f s[2][2] = {};
#pragma unroll
    for (int sc = 0; sc < 2; ++sc) {
      v16h b0 = load_b(ldsK, DH_, sc * 16, 0);
      v16h b1 = load_b(ldsK, DH_, sc * 16, 32);
#pragma unroll
      for (int qs = 0; qs < 2; ++qs) {
        s[qs][sc] = wmma_f16(aq[qs][0], b0, s[qs][sc]);
        s[qs][sc] = wmma_f16(aq[qs][1], b1, s[qs][sc]);
      }
    }

    // ---- online softmax ----
    const bool needmask = (kb == qt);   // only the diagonal block
    float p0[2][8], p1[2][8], alpha[2][8];
#pragma unroll
    for (int qs = 0; qs < 2; ++qs) {
#pragma unroll
      for (int r = 0; r < 8; ++r) {
        const int row = qt * 32 + qs * 16 + r + 8 * halfid;
        float v0 = s[qs][0][r] * scale;
        float v1 = s[qs][1][r] * scale;
        if (needmask) {
          if (tok0 + n      > row) v0 = -3.0e38f;
          if (tok0 + 16 + n > row) v1 = -3.0e38f;
        }
        float bm = fmaxf(v0, v1);
#pragma unroll
        for (int off = 1; off < 16; off <<= 1)
          bm = fmaxf(bm, __shfl_xor(bm, off, 32));
        const float mnew = fmaxf(m_i[qs][r], bm);
        alpha[qs][r] = __expf(m_i[qs][r] - mnew);
        m_i[qs][r] = mnew;
        p0[qs][r] = __expf(v0 - mnew);
        p1[qs][r] = __expf(v1 - mnew);
        float rs = p0[qs][r] + p1[qs][r];
#pragma unroll
        for (int off = 1; off < 16; off <<= 1)
          rs += __shfl_xor(rs, off, 32);
        l_i[qs][r] = l_i[qs][r] * alpha[qs][r] + rs;
      }
    }
#pragma unroll
    for (int qs = 0; qs < 2; ++qs)
#pragma unroll
      for (int nt = 0; nt < 4; ++nt)
#pragma unroll
        for (int r = 0; r < 8; ++r) o[qs][nt][r] *= alpha[qs][r];

    // ---- P: D-layout -> A-layout through LDS ----
    __syncthreads();
#pragma unroll
    for (int qs = 0; qs < 2; ++qs)
#pragma unroll
      for (int r = 0; r < 8; ++r) {
        ldsP[(qs * 16 + r + 8 * halfid) * 32 + n]      = (half_t)p0[qs][r];
        ldsP[(qs * 16 + r + 8 * halfid) * 32 + 16 + n] = (half_t)p1[qs][r];
      }
    __syncthreads();
    const v16h ap0 = load_a(ldsP, 32, 0, 0);
    const v16h ap1 = load_a(ldsP, 32, 16, 0);

    // ---- O += P * V  (8 WMMAs, V B-tiles from LDS reused by both subtiles)
#pragma unroll
    for (int nt = 0; nt < 4; ++nt) {
      v16h bv = load_b(ldsV, 32, nt * 16, 0);
      o[0][nt] = wmma_f16(ap0, bv, o[0][nt]);
      o[1][nt] = wmma_f16(ap1, bv, o[1][nt]);
    }
  }

  // ---- normalize and store ctx (f16, [b,s,d]) ----
#pragma unroll
  for (int qs = 0; qs < 2; ++qs) {
#pragma unroll
    for (int nt = 0; nt < 4; ++nt) {
#pragma unroll
      for (int r = 0; r < 8; ++r) {
        const int srow = qt * 32 + qs * 16 + r + 8 * halfid;
        const float val = o[qs][nt][r] / l_i[qs][r];
        ctx[((size_t)bb * S_ + srow) * D_ + h * DH_ + nt * 16 + n] = (half_t)val;
      }
    }
  }
}

// ---------------------------------------------------------------------------
// Output projection: out = ctx @ Wo^T + bo, f32 output. 32x64 per wave.
// ---------------------------------------------------------------------------
__global__ __launch_bounds__(32) void out_proj_kernel(
    const half_t* __restrict__ ctxh, const half_t* __restrict__ Wh,
    const float* __restrict__ bias, float* __restrict__ out) {
  const int m0 = blockIdx.x * 32;
  const int h  = blockIdx.y;

  v8f acc[2][4] = {};
  for (int kk = 0; kk < D_; kk += 32) {
    v16h a0 = load_a(ctxh, D_, m0,      kk);
    v16h a1 = load_a(ctxh, D_, m0 + 16, kk);
#pragma unroll
    for (int nt = 0; nt < 4; ++nt) {
      v16h b = load_b(Wh, D_, h * DH_ + nt * 16, kk);
      acc[0][nt] = wmma_f16(a0, b, acc[0][nt]);
      acc[1][nt] = wmma_f16(a1, b, acc[1][nt]);
    }
  }

  const int lane   = threadIdx.x & 31;
  const int halfid = lane >> 4;
  const int n      = lane & 15;
#pragma unroll
  for (int mt = 0; mt < 2; ++mt) {
#pragma unroll
    for (int nt = 0; nt < 4; ++nt) {
#pragma unroll
      for (int r = 0; r < 8; ++r) {
        const int flat = m0 + mt * 16 + r + 8 * halfid;
        const int col  = h * DH_ + nt * 16 + n;
        out[(size_t)flat * D_ + col] = acc[mt][nt][r] + bias[col];
      }
    }
  }
}

// ---------------------------------------------------------------------------
// Launch
// ---------------------------------------------------------------------------
extern "C" void kernel_launch(void* const* d_in, const int* in_sizes, int n_in,
                              void* d_out, int out_size, void* d_ws, size_t ws_size,
                              hipStream_t stream) {
  const float* x  = (const float*)d_in[0];
  const float* Wq = (const float*)d_in[1];
  const float* Wk = (const float*)d_in[2];
  const float* Wv = (const float*)d_in[3];
  const float* Wo = (const float*)d_in[4];
  const float* bo = (const float*)d_in[5];
  float* out = (float*)d_out;

  const size_t nX = (size_t)B_ * S_ * D_;   // 3,145,728 elements
  const size_t nW = (size_t)D_ * D_;        //   589,824 elements

  half_t* p    = (half_t*)d_ws;
  half_t* xh   = p; p += nX;
  half_t* Wqh  = p; p += nW;
  half_t* Wkh  = p; p += nW;
  half_t* Wvh  = p; p += nW;
  half_t* Woh  = p; p += nW;
  half_t* Qh   = p; p += nX;
  half_t* Kh   = p; p += nX;
  half_t* Vth  = p; p += nX;
  half_t* ctxh = p; p += nX;   // total ~36.2 MB of workspace

  // 1) precision conversion
  cvt_f32_f16<<<2048, 256, 0, stream>>>(x,  xh,  (int)nX);
  cvt_f32_f16<<<512,  256, 0, stream>>>(Wq, Wqh, (int)nW);
  cvt_f32_f16<<<512,  256, 0, stream>>>(Wk, Wkh, (int)nW);
  cvt_f32_f16<<<512,  256, 0, stream>>>(Wv, Wvh, (int)nW);
  cvt_f32_f16<<<512,  256, 0, stream>>>(Wo, Woh, (int)nW);

  // 2) Q/K/V projections (V stored transposed per head)
  dim3 gProj(B_ * S_ / 32, H_, 1);
  qkv_proj_kernel<<<gProj, 32, 0, stream>>>(xh, Wqh, Qh,  0);
  qkv_proj_kernel<<<gProj, 32, 0, stream>>>(xh, Wkh, Kh,  0);
  qkv_proj_kernel<<<gProj, 32, 0, stream>>>(xh, Wvh, Vth, 1);

  // 3) causal flash attention (async K/V staging through LDS)
  dim3 gAttn(S_ / 32, H_, B_);
  flash_attn_kernel<<<gAttn, 32, 0, stream>>>(Qh, Kh, Vth, ctxh);

  // 4) output projection + bias
  dim3 gOut(B_ * S_ / 32, H_, 1);
  out_proj_kernel<<<gOut, 32, 0, stream>>>(ctxh, Woh, bo, out);
}
